// Mamba_25546465476928
// MI455X (gfx1250) — matrix operand order repaired and checked by reading, not compiled, not run on hardware
//
#include <hip/hip_runtime.h>
#include <hip/hip_bf16.h>

// ---------------------------------------------------------------------------
// Mamba forward (B=1, L=1024, d_model=768, d_inner=1536, n_layer=4, vocab=32000)
// GEMMs via V_WMMA_F32_16X16X32_F16 (f16 in, f32 accumulate).
// Weights pre-transposed to N x Kpad f16; every GEMM N padded to a multiple of
// 64 so the inner loop is branch-free. Each wave computes a 32(M) x 64(N) tile
// (2 A fragments, 8 accumulators, 8 WMMAs per K-step) to halve B-side load
// traffic per FLOP -- the lm_head GEMM is L2-bandwidth-limited on B.
// Fragment loads are address_space(1) v8h loads -> global_load_b128.
// ---------------------------------------------------------------------------

typedef __attribute__((ext_vector_type(16))) _Float16 v16h;
typedef __attribute__((ext_vector_type(8)))  _Float16 v8h;
typedef __attribute__((ext_vector_type(8)))  float    v8f;

typedef __attribute__((address_space(1))) const v8h   gc_v8h;
typedef __attribute__((address_space(1))) const float gc_f32;
typedef __attribute__((address_space(1))) float       g_f32;

#define L_SEQ     1024
#define DMODEL    768
#define DINNER    1536
#define DCONV     4
#define DTRANK    48
#define DTRANKP   64          // dt_rank padded to K multiple of 32
#define DSTATE    16
#define NLAYER    4
#define VOCAB     32000
#define XDBL_PAD  128         // dt_rank + 2*d_state = 80, padded N -> 128

__device__ __forceinline__ float siluf(float x) { return x / (1.f + __expf(-x)); }

// ---------------------------------------------------------------------------
// f32 -> f16, same layout, zero padding outside (srcRows x srcCols).
// ---------------------------------------------------------------------------
__global__ __launch_bounds__(256)
void convert_f16_kernel(const float* __restrict__ src, int srcLd, int srcRows,
                        int srcCols, _Float16* __restrict__ dst, int dstLd,
                        int total)
{
    int i = blockIdx.x * 256 + threadIdx.x;
    if (i >= total) return;
    int r = i / dstLd, c = i % dstLd;
    float v = (r < srcRows && c < srcCols) ? src[(size_t)r * srcLd + c] : 0.f;
    dst[i] = (_Float16)v;
}

// ---------------------------------------------------------------------------
// f32 (K x N row-major) -> f16 transposed (Npad x Kpad row-major),
// zero pad outside srcK x srcN. total = Npad * Kpad.
// ---------------------------------------------------------------------------
__global__ __launch_bounds__(256)
void convert_f16_t_kernel(const float* __restrict__ src, int srcK, int srcN,
                          _Float16* __restrict__ dst, int Kpad, int total)
{
    int i = blockIdx.x * 256 + threadIdx.x;
    if (i >= total) return;
    int n = i / Kpad, k = i % Kpad;
    float v = (k < srcK && n < srcN) ? src[(size_t)k * srcN + n] : 0.f;
    dst[i] = (_Float16)v;
}

// ---------------------------------------------------------------------------
// Embedding gather: x[t, :] = emb[ids[t], :]
// ---------------------------------------------------------------------------
__global__ __launch_bounds__(256)
void embed_kernel(const int* __restrict__ ids, const float* __restrict__ emb,
                  float* __restrict__ x)
{
    int i = blockIdx.x * 256 + threadIdx.x;
    if (i >= L_SEQ * DMODEL) return;
    int t = i / DMODEL, c = i % DMODEL;
    x[i] = emb[(size_t)ids[t] * DMODEL + c];
}

// ---------------------------------------------------------------------------
// RMSNorm of one row per block; writes f16 for WMMA A operand.
// ---------------------------------------------------------------------------
__global__ __launch_bounds__(256)
void rmsnorm_f16_kernel(const float* __restrict__ x, const float* __restrict__ w,
                        _Float16* __restrict__ out)
{
    __shared__ float red[256];
    const int row = blockIdx.x;
    const float* xr = x + (size_t)row * DMODEL;
    float s = 0.f;
    for (int c = threadIdx.x; c < DMODEL; c += 256) { float v = xr[c]; s += v * v; }
    red[threadIdx.x] = s;
    __syncthreads();
    for (int off = 128; off > 0; off >>= 1) {
        if (threadIdx.x < off) red[threadIdx.x] += red[threadIdx.x + off];
        __syncthreads();
    }
    const float rs = rsqrtf(red[0] / (float)DMODEL + 1e-4f);
    for (int c = threadIdx.x; c < DMODEL; c += 256)
        out[(size_t)row * DMODEL + c] = (_Float16)(xr[c] * rs * w[c]);
}

// ---------------------------------------------------------------------------
// WMMA GEMM: C[M,N] (f32) = A[M,K] (f16 row-major, lda) x Bt^T
//   Bt is N x K row-major (ldb = K stride) -- weights pre-transposed.
// HAS_RESID: accumulator initialized from Resid (fused residual add via C op).
// One wave computes a 32(M) x 64(N) tile; 4 waves/block -> 128x64 block tile.
// Per K-step: 4 A loads + 8 B loads (b128) feed 8 WMMAs (B reused across the
// two M halves). M multiple of 128; N multiple of 64; K multiple of 32.
// Inner loop is branch-free; EXEC all-ones at every WMMA (ISA 7.12 req).
// ---------------------------------------------------------------------------
template<bool HAS_RESID>
__global__ __launch_bounds__(128)
void gemm_wmma_f16_kernel(const _Float16* __restrict__ A,
                          const _Float16* __restrict__ Bt,
                          const float* __restrict__ Resid,
                          float* __restrict__ C,
                          int M, int N, int K, int lda, int ldb, int ldc)
{
    const int lane = threadIdx.x & 31;
    const int wave = threadIdx.x >> 5;
    const int m0 = (blockIdx.y * 4 + wave) * 32;
    const int n0 = blockIdx.x * 64;
    if (m0 >= M || n0 >= N) return;

    const int half = lane >> 4;          // half-wave select
    const int lnm  = lane & 15;

    // A: lane = row; half 0 holds K 0-7 & 16-23 of each 32-slab,
    // half 1 holds K 8-15 & 24-31 (ISA 7.12.2 16-bit A layout).
    gc_v8h* Ag0 = (gc_v8h*)(A + (size_t)(m0 + lnm) * lda + half * 8);
    gc_v8h* Ag1 = (gc_v8h*)(A + (size_t)(m0 + 16 + lnm) * lda + half * 8);
    // B: lane = column n+lnm; half 0 holds K 0-7 & 16-23, half 1 K 8-15 & 24-31.
    gc_v8h* Bg[4];
#pragma unroll
    for (int j = 0; j < 4; ++j)
        Bg[j] = (gc_v8h*)(Bt + (size_t)(n0 + j * 16 + lnm) * ldb + half * 8);
    const _Float16* apre0 = A + (size_t)(m0 + lnm) * lda + half * 8;
    const _Float16* apre1 = A + (size_t)(m0 + 16 + lnm) * lda + half * 8;

    v8f acc0[4], acc1[4];
#pragma unroll
    for (int j = 0; j < 4; ++j) {
        if (HAS_RESID) {
            gc_f32* rp0 = (gc_f32*)(Resid + (size_t)(m0 + half * 8) * ldc
                                    + n0 + j * 16 + lnm);
            gc_f32* rp1 = (gc_f32*)(Resid + (size_t)(m0 + 16 + half * 8) * ldc
                                    + n0 + j * 16 + lnm);
#pragma unroll
            for (int v = 0; v < 8; ++v) {
                acc0[j][v] = rp0[(size_t)v * ldc];
                acc1[j][v] = rp1[(size_t)v * ldc];
            }
        } else {
#pragma unroll
            for (int v = 0; v < 8; ++v) { acc0[j][v] = 0.f; acc1[j][v] = 0.f; }
        }
    }

    int av = 0;   // v8h-unit offset into Ag* (k0/8)
    int bv = 0;   // v8h-unit offset into Bg[j]
    for (int k0 = 0; k0 < K; k0 += 32) {
        v8h a0lo = Ag0[av];          // global_load_b128 x4 (two A fragments)
        v8h a0hi = Ag0[av + 2];
        v8h a1lo = Ag1[av];
        v8h a1hi = Ag1[av + 2];
        v16h a0 = __builtin_shufflevector(a0lo, a0hi,
                  0, 1, 2, 3, 4, 5, 6, 7, 8, 9, 10, 11, 12, 13, 14, 15);
        v16h a1 = __builtin_shufflevector(a1lo, a1hi,
                  0, 1, 2, 3, 4, 5, 6, 7, 8, 9, 10, 11, 12, 13, 14, 15);
        if (k0 + 64 < K) {
            __builtin_prefetch(apre0 + k0 + 64, 0, 3);   // global_prefetch_b8
            __builtin_prefetch(apre1 + k0 + 64, 0, 3);
        }
        av += 4;

#pragma unroll
        for (int j = 0; j < 4; ++j) {
            v8h blo = Bg[j][bv];     // 2x global_load_b128
            v8h bhi = Bg[j][bv + 2];
            v16h b = __builtin_shufflevector(blo, bhi,
                     0, 1, 2, 3, 4, 5, 6, 7, 8, 9, 10, 11, 12, 13, 14, 15);
            acc0[j] = __builtin_amdgcn_wmma_f32_16x16x32_f16(
                false, a0, false, b, (short)0, acc0[j], false, false);
            acc1[j] = __builtin_amdgcn_wmma_f32_16x16x32_f16(
                false, a1, false, b, (short)0, acc1[j], false, false);
        }
        bv += 4;
    }

#pragma unroll
    for (int j = 0; j < 4; ++j) {
        g_f32* cp0 = (g_f32*)(C + (size_t)(m0 + half * 8) * ldc + n0 + j * 16 + lnm);
        g_f32* cp1 = (g_f32*)(C + (size_t)(m0 + 16 + half * 8) * ldc + n0 + j * 16 + lnm);
#pragma unroll
        for (int v = 0; v < 8; ++v) {
            cp0[(size_t)v * ldc] = acc0[j][v];
            cp1[(size_t)v * ldc] = acc1[j][v];
        }
    }
}

// ---------------------------------------------------------------------------
// Depthwise causal conv (width 4) + bias + SiLU.
// Input = first half of xr rows (1024 x 3072), output xi (1024 x 1536).
// ---------------------------------------------------------------------------
__global__ __launch_bounds__(256)
void conv_silu_kernel(const float* __restrict__ xr, const float* __restrict__ cw,
                      const float* __restrict__ cb, float* __restrict__ xi)
{
    int i = blockIdx.x * 256 + threadIdx.x;
    if (i >= L_SEQ * DINNER) return;
    int t = i / DINNER, c = i % DINNER;
    float s = cb[c];
#pragma unroll
    for (int j = 0; j < DCONV; ++j) {
        int tt = t - (DCONV - 1) + j;
        if (tt >= 0) s += cw[j * DINNER + c] * xr[(size_t)tt * (2 * DINNER) + c];
    }
    xi[i] = siluf(s);
}

// delta = softplus(delta_raw + dt_b), in place (1024 x 1536)
__global__ __launch_bounds__(256)
void softplus_bias_kernel(float* __restrict__ delta, const float* __restrict__ dt_b)
{
    int i = blockIdx.x * 256 + threadIdx.x;
    if (i >= L_SEQ * DINNER) return;
    float v = delta[i] + dt_b[i % DINNER];
    delta[i] = (v > 20.f) ? v : log1pf(__expf(v));
}

// ---------------------------------------------------------------------------
// Selective scan: one thread per d_inner channel, 16-state recurrence in
// registers; B/C rows staged through LDS in 64-step chunks (broadcast reads).
// ---------------------------------------------------------------------------
__global__ __launch_bounds__(256)
void scan_kernel(const float* __restrict__ u, const float* __restrict__ delta,
                 const float* __restrict__ xdbl, const float* __restrict__ A_log,
                 const float* __restrict__ Dp, float* __restrict__ y)
{
    __shared__ float lbc[64 * 32];
    const int d = blockIdx.x * 256 + threadIdx.x;   // 0..1535, exact
    float An[DSTATE], h[DSTATE];
#pragma unroll
    for (int n = 0; n < DSTATE; ++n) {
        An[n] = -__expf(A_log[(size_t)d * DSTATE + n]);
        h[n] = 0.f;
    }
    const float Dd = Dp[d];

    for (int t0 = 0; t0 < L_SEQ; t0 += 64) {
        __syncthreads();
        for (int i = threadIdx.x; i < 64 * 32; i += 256) {
            int tt = i >> 5, e = i & 31;
            lbc[i] = xdbl[(size_t)(t0 + tt) * XDBL_PAD + DTRANK + e];
        }
        __syncthreads();
        for (int tt = 0; tt < 64; ++tt) {
            const int t = t0 + tt;
            const float dt = delta[(size_t)t * DINNER + d];
            const float uv = u[(size_t)t * DINNER + d];
            const float du = dt * uv;
            const float* bc = &lbc[tt * 32];
            float yv = 0.f;
#pragma unroll
            for (int n = 0; n < DSTATE; ++n) {
                h[n] = __expf(dt * An[n]) * h[n] + du * bc[n];
                yv += h[n] * bc[16 + n];
            }
            y[(size_t)t * DINNER + d] = yv + uv * Dd;
        }
    }
}

// z16 = f16( y * silu(res) ), res = second half of xr rows
__global__ __launch_bounds__(256)
void gate_f16_kernel(const float* __restrict__ y, const float* __restrict__ xr,
                     _Float16* __restrict__ z16)
{
    int i = blockIdx.x * 256 + threadIdx.x;
    if (i >= L_SEQ * DINNER) return;
    int t = i / DINNER, c = i % DINNER;
    float r = xr[(size_t)t * (2 * DINNER) + DINNER + c];
    z16[i] = (_Float16)(y[i] * siluf(r));
}

// ---------------------------------------------------------------------------
extern "C" void kernel_launch(void* const* d_in, const int* in_sizes, int n_in,
                              void* d_out, int out_size, void* d_ws, size_t ws_size,
                              hipStream_t stream)
{
    (void)in_sizes; (void)n_in; (void)out_size; (void)ws_size;

    const int*   ids      = (const int*)  d_in[0];
    const float* emb      = (const float*)d_in[1];
    const float* norm_ws  = (const float*)d_in[2];
    const float* in_ws    = (const float*)d_in[3];
    const float* conv_ws  = (const float*)d_in[4];
    const float* conv_bs  = (const float*)d_in[5];
    const float* xproj_ws = (const float*)d_in[6];
    const float* dt_ws    = (const float*)d_in[7];
    const float* dt_bs    = (const float*)d_in[8];
    const float* A_logs   = (const float*)d_in[9];
    const float* Ds       = (const float*)d_in[10];
    const float* out_wsp  = (const float*)d_in[11];
    const float* norm_f_w = (const float*)d_in[12];
    float* out = (float*)d_out;

    // --- workspace carve-up (~100 MB total) ---
    char* base = (char*)d_ws;
    size_t off = 0;
    auto alloc = [&](size_t bytes) -> void* {
        void* p = base + off;
        off += (bytes + 255) & ~(size_t)255;
        return p;
    };
    _Float16* emb16   = (_Float16*)alloc((size_t)VOCAB * DMODEL * 2);   // N x K already
    float*    x       = (float*)   alloc((size_t)L_SEQ * DMODEL * 4);
    _Float16* xn16    = (_Float16*)alloc((size_t)L_SEQ * DMODEL * 2);
    float*    xr      = (float*)   alloc((size_t)L_SEQ * 2 * DINNER * 4);
    float*    xi      = (float*)   alloc((size_t)L_SEQ * DINNER * 4);
    _Float16* xi16    = (_Float16*)alloc((size_t)L_SEQ * DINNER * 2);
    float*    xdbl    = (float*)   alloc((size_t)L_SEQ * XDBL_PAD * 4);
    _Float16* dtA16   = (_Float16*)alloc((size_t)L_SEQ * DTRANKP * 2);
    float*    delta   = (float*)   alloc((size_t)L_SEQ * DINNER * 4);
    float*    yb      = (float*)   alloc((size_t)L_SEQ * DINNER * 4);
    _Float16* z16     = (_Float16*)alloc((size_t)L_SEQ * DINNER * 2);
    _Float16* inw16t  = (_Float16*)alloc((size_t)(2 * DINNER) * DMODEL * 2); // 3072 x 768
    _Float16* xprojw16t=(_Float16*)alloc((size_t)XDBL_PAD * DINNER * 2);     // 128 x 1536
    _Float16* dtw16t  = (_Float16*)alloc((size_t)DINNER * DTRANKP * 2);      // 1536 x 64
    _Float16* outw16t = (_Float16*)alloc((size_t)DMODEL * DINNER * 2);       // 768 x 1536

    const int LI = L_SEQ * DINNER;                 // 1,572,864
    const int LD = L_SEQ * DMODEL;                 // 786,432
    auto blocks = [](int n) { return (n + 255) / 256; };

    // Embedding table -> f16 once (49 MB; fits in the 192 MB L2 for lm_head reuse)
    {
        int tot = VOCAB * DMODEL;
        convert_f16_kernel<<<blocks(tot), 256, 0, stream>>>(
            emb, DMODEL, VOCAB, DMODEL, emb16, DMODEL, tot);
    }
    // x = embedding[ids]
    embed_kernel<<<blocks(LD), 256, 0, stream>>>(ids, emb, x);

    for (int l = 0; l < NLAYER; ++l) {
        // RMSNorm -> f16 activations
        rmsnorm_f16_kernel<<<L_SEQ, 256, 0, stream>>>(x, norm_ws + l * DMODEL, xn16);

        // in_proj: xr(1024x3072) = xn16(1024x768) @ in_w(768x3072)
        {
            int tot = 2 * DINNER * DMODEL;
            convert_f16_t_kernel<<<blocks(tot), 256, 0, stream>>>(
                in_ws + (size_t)l * DMODEL * 2 * DINNER, DMODEL, 2 * DINNER,
                inw16t, DMODEL, tot);
            dim3 g((2 * DINNER) / 64, L_SEQ / 128);
            gemm_wmma_f16_kernel<false><<<g, 128, 0, stream>>>(
                xn16, inw16t, nullptr, xr,
                L_SEQ, 2 * DINNER, DMODEL, DMODEL, DMODEL, 2 * DINNER);
        }

        // depthwise causal conv + SiLU on first half of xr
        conv_silu_kernel<<<blocks(LI), 256, 0, stream>>>(
            xr, conv_ws + (size_t)l * DCONV * DINNER, conv_bs + l * DINNER, xi);

        // x_proj: xdbl(1024x128, cols 80..127 zero) = xi(1024x1536) @ xproj_w
        {
            convert_f16_kernel<<<blocks(LI), 256, 0, stream>>>(
                xi, DINNER, L_SEQ, DINNER, xi16, DINNER, LI);
            int tot = XDBL_PAD * DINNER;                   // Npad=128 x K=1536
            convert_f16_t_kernel<<<blocks(tot), 256, 0, stream>>>(
                xproj_ws + (size_t)l * DINNER * 80, DINNER, 80,
                xprojw16t, DINNER, tot);
            dim3 g(XDBL_PAD / 64, L_SEQ / 128);
            gemm_wmma_f16_kernel<false><<<g, 128, 0, stream>>>(
                xi16, xprojw16t, nullptr, xdbl,
                L_SEQ, XDBL_PAD, DINNER, DINNER, DINNER, XDBL_PAD);
        }

        // dt_proj: delta_raw(1024x1536) = xdbl[:, :48](pad->64) @ dt_w(48->64 x 1536)
        {
            convert_f16_kernel<<<blocks(L_SEQ * DTRANKP), 256, 0, stream>>>(
                xdbl, XDBL_PAD, L_SEQ, DTRANK, dtA16, DTRANKP, L_SEQ * DTRANKP);
            convert_f16_t_kernel<<<blocks(DINNER * DTRANKP), 256, 0, stream>>>(
                dt_ws + (size_t)l * DTRANK * DINNER, DTRANK, DINNER,
                dtw16t, DTRANKP, DINNER * DTRANKP);
            dim3 g(DINNER / 64, L_SEQ / 128);
            gemm_wmma_f16_kernel<false><<<g, 128, 0, stream>>>(
                dtA16, dtw16t, nullptr, delta,
                L_SEQ, DINNER, DTRANKP, DTRANKP, DTRANKP, DINNER);
            softplus_bias_kernel<<<blocks(LI), 256, 0, stream>>>(
                delta, dt_bs + l * DINNER);
        }

        // selective scan (sequential in t, parallel over 1536 channels)
        scan_kernel<<<DINNER / 256, 256, 0, stream>>>(
            xi, delta, xdbl, A_logs + (size_t)l * DINNER * DSTATE,
            Ds + l * DINNER, yb);

        // gate + out_proj with fused residual: x += (y*silu(res)) @ out_w
        {
            gate_f16_kernel<<<blocks(LI), 256, 0, stream>>>(yb, xr, z16);
            convert_f16_t_kernel<<<blocks(DMODEL * DINNER), 256, 0, stream>>>(
                out_wsp + (size_t)l * DINNER * DMODEL, DINNER, DMODEL,
                outw16t, DINNER, DMODEL * DINNER);
            dim3 g(DMODEL / 64, L_SEQ / 128);
            gemm_wmma_f16_kernel<true><<<g, 128, 0, stream>>>(
                z16, outw16t, /*Resid=*/x, /*C=*/x,
                L_SEQ, DMODEL, DINNER, DINNER, DINNER, DMODEL);
        }
    }

    // final RMSNorm, then tied lm_head: logits = xf @ emb^T (emb16 is N x K)
    rmsnorm_f16_kernel<<<L_SEQ, 256, 0, stream>>>(x, norm_f_w, xn16);
    {
        dim3 g(VOCAB / 64, L_SEQ / 128);
        gemm_wmma_f16_kernel<false><<<g, 128, 0, stream>>>(
            xn16, emb16, nullptr, out,
            L_SEQ, VOCAB, DMODEL, DMODEL, DMODEL, VOCAB);
    }
}
